// Decoder_90709709292135
// MI455X (gfx1250) — matrix operand order repaired and checked by reading, not compiled, not run on hardware
//
#include <hip/hip_runtime.h>
#include <hip/hip_bf16.h>

typedef unsigned short ushort_t;
typedef __attribute__((ext_vector_type(16))) __bf16 v16bf;
typedef __attribute__((ext_vector_type(8)))  float  v8f;

#define B_  32
#define S_  50
#define H_  1024
#define V_  32000
#define DW_ 512
#define T_  32
#define DIN_ 1536

// ---------- helpers ----------

__device__ __forceinline__ ushort_t f2bf(float f) {
  union { float f; unsigned int u; } x; x.f = f;
  unsigned int u = x.u + 0x7FFFu + ((x.u >> 16) & 1u);   // round-to-nearest-even
  return (ushort_t)(u >> 16);
}

// A fragment: 16x32 bf16 tile, row-major, per CDNA5 ISA layout:
// lanes 0-15 : row m=lane,    elems 0-7 = K[0..7],  elems 8-15 = K[16..23]
// lanes 16-31: row m=lane-16, elems 0-7 = K[8..15], elems 8-15 = K[24..31]
__device__ __forceinline__ v16bf load_a_frag(const ushort_t* __restrict__ tile, int ld, int lane) {
  const int r  = lane & 15;
  const int ko = (lane >> 4) << 3;            // 0 or 8
  union { uint4 u[2]; v16bf v; } x;
  const ushort_t* p = tile + (size_t)r * ld;
  x.u[0] = *(const uint4*)(p + ko);
  x.u[1] = *(const uint4*)(p + 16 + ko);
  return x.v;
}

// B fragment: B is KxN; we hold B^T (the weight matrix, N x K row-major).
// Per ISA: lane n=lane&15 holds column n; lanes 0-15 -> K[0..7],K[16..23], lanes 16-31 -> K[8..15],K[24..31].
__device__ __forceinline__ v16bf load_b_frag(const ushort_t* __restrict__ bt_tile, int ld, int lane) {
  const int n  = lane & 15;
  const int kb = (lane >> 4) << 3;            // 0 or 8
  union { uint4 u[2]; v16bf v; } x;
  const ushort_t* p = bt_tile + (size_t)n * ld + kb;
  x.u[0] = *(const uint4*)(p);
  x.u[1] = *(const uint4*)(p + 16);
  return x.v;
}

// ---------- GEMM: C[M,N] = A[M,K] * Bt[N,K]^T + bias ----------
// One wave computes a 16x64 strip (4 WMMAs / 32-wide K step, A-frag reused).
// Software-pipelined: fragments for k+32 are loaded before the WMMAs for k,
// so the compiler can issue partial s_wait_loadcnt and overlap VMEM with the
// matrix pipe instead of draining loadcnt to 0 before every WMMA.
__global__ void gemm_bt_wmma(const ushort_t* __restrict__ A,
                             const ushort_t* __restrict__ Bt,
                             const float*    __restrict__ bias,
                             float*          __restrict__ C,
                             int M, int N, int K, int lda, int ldb, long ldc) {
  const int wave = blockIdx.x * 8 + (threadIdx.x >> 5);
  const int lane = threadIdx.x & 31;
  const int mtiles = M >> 4;
  const int ntiles = N >> 6;
  if (wave >= mtiles * ntiles) return;
  const int mt = wave % mtiles;
  const int nt = wave / mtiles;
  const int n0 = nt << 6;

  const ushort_t* Arow = A + (size_t)(mt << 4) * lda;
  const ushort_t* Brow0 = Bt + (size_t)(n0 +  0) * ldb;
  const ushort_t* Brow1 = Bt + (size_t)(n0 + 16) * ldb;
  const ushort_t* Brow2 = Bt + (size_t)(n0 + 32) * ldb;
  const ushort_t* Brow3 = Bt + (size_t)(n0 + 48) * ldb;

  v8f acc[4] = {};

  // prologue: fragments for k = 0
  v16bf a  = load_a_frag(Arow, lda, lane);
  v16bf b0 = load_b_frag(Brow0, ldb, lane);
  v16bf b1 = load_b_frag(Brow1, ldb, lane);
  v16bf b2 = load_b_frag(Brow2, ldb, lane);
  v16bf b3 = load_b_frag(Brow3, ldb, lane);

  for (int k = 32; k < K; k += 32) {
    // prefetch next K-slice before consuming the current one
    v16bf an  = load_a_frag(Arow + k, lda, lane);
    v16bf bn0 = load_b_frag(Brow0 + k, ldb, lane);
    v16bf bn1 = load_b_frag(Brow1 + k, ldb, lane);
    v16bf bn2 = load_b_frag(Brow2 + k, ldb, lane);
    v16bf bn3 = load_b_frag(Brow3 + k, ldb, lane);

    acc[0] = __builtin_amdgcn_wmma_f32_16x16x32_bf16(false, a, false, b0, (short)0, acc[0], false, false);
    acc[1] = __builtin_amdgcn_wmma_f32_16x16x32_bf16(false, a, false, b1, (short)0, acc[1], false, false);
    acc[2] = __builtin_amdgcn_wmma_f32_16x16x32_bf16(false, a, false, b2, (short)0, acc[2], false, false);
    acc[3] = __builtin_amdgcn_wmma_f32_16x16x32_bf16(false, a, false, b3, (short)0, acc[3], false, false);

    a = an; b0 = bn0; b1 = bn1; b2 = bn2; b3 = bn3;
  }
  // epilogue: last K-slice
  acc[0] = __builtin_amdgcn_wmma_f32_16x16x32_bf16(false, a, false, b0, (short)0, acc[0], false, false);
  acc[1] = __builtin_amdgcn_wmma_f32_16x16x32_bf16(false, a, false, b1, (short)0, acc[1], false, false);
  acc[2] = __builtin_amdgcn_wmma_f32_16x16x32_bf16(false, a, false, b2, (short)0, acc[2], false, false);
  acc[3] = __builtin_amdgcn_wmma_f32_16x16x32_bf16(false, a, false, b3, (short)0, acc[3], false, false);

  // C/D layout: VGPR i -> row = i + 8*(lane>=16), col = lane&15
  const int row0 = (mt << 4) + ((lane >> 4) << 3);
  const int nc   = lane & 15;
  #pragma unroll
  for (int j = 0; j < 4; ++j) {
    const int col = n0 + (j << 4) + nc;
    const float bv = bias ? bias[col] : 0.0f;
    #pragma unroll
    for (int i = 0; i < 8; ++i)
      C[(size_t)(row0 + i) * (size_t)ldc + col] = acc[j][i] + bv;
  }
}

// ---------- setup kernels ----------

__global__ void k_f32_to_bf16(const float* __restrict__ s, ushort_t* __restrict__ d, long n) {
  long i = (long)blockIdx.x * blockDim.x + threadIdx.x;
  const long stride = (long)gridDim.x * blockDim.x;
  for (; i < n; i += stride) d[i] = f2bf(s[i]);
}

// W1 is [H, 2H]: cols [0,H) hit enc part of hcat, cols [H,2H) hit h part.
__global__ void k_split_w1(const float* __restrict__ W1, ushort_t* e, ushort_t* h) {
  int i = blockIdx.x * blockDim.x + threadIdx.x;
  if (i >= H_ * H_) return;
  int k = i >> 10, d = i & 1023;
  e[i] = f2bf(W1[(size_t)k * 2048 + d]);
  h[i] = f2bf(W1[(size_t)k * 2048 + 1024 + d]);
}

__global__ void k_init_h(const float* __restrict__ eh, float* h, ushort_t* hbf) {
  int i = blockIdx.x * blockDim.x + threadIdx.x;
  if (i < B_ * H_) { float v = eh[i]; h[i] = v; hbf[i] = f2bf(v); }
}

// ---------- attention scores: e[b,s] = sum_h tanh(pre_enc + pre_h) * w2[h] ----------
__global__ void k_scores(const float* __restrict__ pe, const float* __restrict__ ph,
                         const float* __restrict__ w2, float* __restrict__ e) {
  __shared__ float sm[8];
  const int bs = blockIdx.x;
  const int b  = bs / S_;
  const float* p0 = pe + (size_t)bs * H_;
  const float* p1 = ph + (size_t)b * H_;
  float acc = 0.f;
  for (int h = threadIdx.x; h < H_; h += 256)
    acc += tanhf(p0[h] + p1[h]) * w2[h];
  #pragma unroll
  for (int o = 16; o; o >>= 1) acc += __shfl_down(acc, o, 32);
  if ((threadIdx.x & 31) == 0) sm[threadIdx.x >> 5] = acc;
  __syncthreads();
  if (threadIdx.x == 0) {
    float v = 0.f;
    #pragma unroll
    for (int i = 0; i < 8; ++i) v += sm[i];
    e[bs] = v;
  }
}

// ---------- softmax over S + context + build dec_in (bf16) ----------
__global__ void k_softmax_ctx(const float* __restrict__ e, const float* __restrict__ enc,
                              const float* __restrict__ emb, const int* __restrict__ tgt,
                              int t, ushort_t* __restrict__ dec_in) {
  __shared__ float alpha[S_];
  const int b = blockIdx.x;
  if (threadIdx.x == 0) {
    float mx = -3.0e38f;
    for (int s = 0; s < S_; ++s) { float v = e[b * S_ + s]; mx = v > mx ? v : mx; }
    float sum = 0.f;
    for (int s = 0; s < S_; ++s) { float v = expf(e[b * S_ + s] - mx); alpha[s] = v; sum += v; }
    const float inv = 1.f / sum;
    for (int s = 0; s < S_; ++s) alpha[s] *= inv;
  }
  __syncthreads();
  // x_t = emb[targets[b, t]]
  const int tok = tgt[b * T_ + t];
  for (int j = threadIdx.x; j < DW_; j += 256)
    dec_in[b * DIN_ + j] = f2bf(emb[(size_t)tok * DW_ + j]);
  // context = alpha @ enc[b]
  for (int h = threadIdx.x; h < H_; h += 256) {
    float acc = 0.f;
    const float* ep = enc + (size_t)b * S_ * H_ + h;
    #pragma unroll 5
    for (int s = 0; s < S_; ++s) acc += alpha[s] * ep[(size_t)s * H_];
    dec_in[b * DIN_ + DW_ + h] = f2bf(acc);
  }
}

// ---------- GRU gate fusion ----------
__global__ void k_gates(const float* __restrict__ gi, const float* __restrict__ gh,
                        float* __restrict__ h, ushort_t* __restrict__ hbf) {
  int i = blockIdx.x * blockDim.x + threadIdx.x;
  if (i >= B_ * H_) return;
  const int b = i >> 10, hh = i & 1023;
  const float* gib = gi + (size_t)b * 3 * H_;
  const float* ghb = gh + (size_t)b * 3 * H_;
  const float r = 1.f / (1.f + expf(-(gib[hh] + ghb[hh])));
  const float z = 1.f / (1.f + expf(-(gib[H_ + hh] + ghb[H_ + hh])));
  const float n = tanhf(gib[2 * H_ + hh] + r * ghb[2 * H_ + hh]);
  const float hn = (1.f - z) * n + z * h[i];
  h[i] = hn;
  hbf[i] = f2bf(hn);
}

// ---------- in-place log_softmax over V per row ----------
__global__ void k_logsoftmax(float* __restrict__ P, long rowstride) {
  __shared__ float sm[8];
  __shared__ float bc;
  float* row = P + (size_t)blockIdx.x * (size_t)rowstride;
  const int lane = threadIdx.x & 31, w = threadIdx.x >> 5;

  float mx = -3.0e38f;
  for (int v = threadIdx.x; v < V_; v += 256) mx = fmaxf(mx, row[v]);
  #pragma unroll
  for (int o = 16; o; o >>= 1) mx = fmaxf(mx, __shfl_down(mx, o, 32));
  if (lane == 0) sm[w] = mx;
  __syncthreads();
  if (threadIdx.x == 0) {
    float v = sm[0];
    #pragma unroll
    for (int i = 1; i < 8; ++i) v = fmaxf(v, sm[i]);
    bc = v;
  }
  __syncthreads();
  mx = bc;

  float sum = 0.f;
  for (int v = threadIdx.x; v < V_; v += 256) sum += expf(row[v] - mx);
  #pragma unroll
  for (int o = 16; o; o >>= 1) sum += __shfl_down(sum, o, 32);
  __syncthreads();
  if (lane == 0) sm[w] = sum;
  __syncthreads();
  if (threadIdx.x == 0) {
    float v = 0.f;
    #pragma unroll
    for (int i = 0; i < 8; ++i) v += sm[i];
    bc = mx + logf(v);
  }
  __syncthreads();
  const float lz = bc;
  for (int v = threadIdx.x; v < V_; v += 256) row[v] = row[v] - lz;
}

// ---------- launch ----------
extern "C" void kernel_launch(void* const* d_in, const int* in_sizes, int n_in,
                              void* d_out, int out_size, void* d_ws, size_t ws_size,
                              hipStream_t stream) {
  const float* enc   = (const float*)d_in[0];   // [B,S,H]
  const float* ehid  = (const float*)d_in[1];   // [1,B,H]
  const int*   tgt   = (const int*)  d_in[2];   // [B,T]
  const float* emb   = (const float*)d_in[3];   // [32000,512]
  const float* W1    = (const float*)d_in[4];   // [H,2H]
  const float* b1    = (const float*)d_in[5];   // [H]
  const float* W2    = (const float*)d_in[6];   // [1,H]
  const float* W_ih  = (const float*)d_in[7];   // [3H,Din]
  const float* b_ih  = (const float*)d_in[8];
  const float* W_hh  = (const float*)d_in[9];   // [3H,H]
  const float* b_hh  = (const float*)d_in[10];
  const float* W_out = (const float*)d_in[11];  // [V,H]
  const float* b_out = (const float*)d_in[12];
  float* out = (float*)d_out;                   // [B, T-1, V]

  char* ws = (char*)d_ws;
  size_t off = 0;
  auto alloc = [&](size_t bytes) -> char* {
    char* p = ws + off;
    off = (off + bytes + 255) & ~(size_t)255;
    return p;
  };
  ushort_t* wout_bf = (ushort_t*)alloc((size_t)V_ * H_ * 2);       // 65.5 MB
  ushort_t* wih_bf  = (ushort_t*)alloc((size_t)3 * H_ * DIN_ * 2); //  9.4 MB
  ushort_t* whh_bf  = (ushort_t*)alloc((size_t)3 * H_ * H_ * 2);   //  6.3 MB
  ushort_t* w1e_bf  = (ushort_t*)alloc((size_t)H_ * H_ * 2);
  ushort_t* w1h_bf  = (ushort_t*)alloc((size_t)H_ * H_ * 2);
  ushort_t* enc_bf  = (ushort_t*)alloc((size_t)B_ * S_ * H_ * 2);
  float*    pre_enc = (float*)   alloc((size_t)B_ * S_ * H_ * 4);  //  6.5 MB
  float*    pre_h   = (float*)   alloc((size_t)B_ * H_ * 4);
  float*    hbuf    = (float*)   alloc((size_t)B_ * H_ * 4);
  ushort_t* hbf     = (ushort_t*)alloc((size_t)B_ * H_ * 2);
  float*    ebuf    = (float*)   alloc((size_t)B_ * S_ * 4);
  ushort_t* dec_in  = (ushort_t*)alloc((size_t)B_ * DIN_ * 2);
  float*    gi      = (float*)   alloc((size_t)B_ * 3 * H_ * 4);
  float*    gh      = (float*)   alloc((size_t)B_ * 3 * H_ * 4);

  // One-time: bf16 weight staging (then resident in 192MB L2 across the scan)
  k_f32_to_bf16<<<1024, 256, 0, stream>>>(W_out, wout_bf, (long)V_ * H_);
  k_f32_to_bf16<<<512, 256, 0, stream>>>(W_ih, wih_bf, (long)3 * H_ * DIN_);
  k_f32_to_bf16<<<512, 256, 0, stream>>>(W_hh, whh_bf, (long)3 * H_ * H_);
  k_f32_to_bf16<<<512, 256, 0, stream>>>(enc, enc_bf, (long)B_ * S_ * H_);
  k_split_w1<<<(H_ * H_ + 255) / 256, 256, 0, stream>>>(W1, w1e_bf, w1h_bf);
  k_init_h<<<(B_ * H_ + 255) / 256, 256, 0, stream>>>(ehid, hbuf, hbf);

  auto gemm = [&](const ushort_t* A, const ushort_t* Bt, const float* bias, float* C,
                  int M, int N, int K, int lda, int ldb, long ldc) {
    const int waves = (M >> 4) * (N >> 6);
    gemm_bt_wmma<<<(waves + 7) / 8, 256, 0, stream>>>(A, Bt, bias, C, M, N, K, lda, ldb, ldc);
  };

  // Hoisted, step-invariant half of the attention projection: enc @ W1e^T + b1
  gemm(enc_bf, w1e_bf, b1, pre_enc, B_ * S_, H_, H_, H_, H_, H_);

  for (int t = 0; t < T_ - 1; ++t) {
    // pre_h = h @ W1h^T   (the only attention GEMM that changes per step)
    gemm(hbf, w1h_bf, nullptr, pre_h, B_, H_, H_, H_, H_, H_);
    k_scores<<<B_ * S_, 256, 0, stream>>>(pre_enc, pre_h, W2, ebuf);
    k_softmax_ctx<<<B_, 256, 0, stream>>>(ebuf, enc, emb, tgt, t, dec_in);
    // GRU gate GEMMs
    gemm(dec_in, wih_bf, b_ih, gi, B_, 3 * H_, DIN_, DIN_, DIN_, 3 * H_);
    gemm(hbf, whh_bf, b_hh, gh, B_, 3 * H_, H_, H_, H_, 3 * H_);
    k_gates<<<(B_ * H_ + 255) / 256, 256, 0, stream>>>(gi, gh, hbuf, hbf);
    // Output projection straight into d_out[:, t, :]
    float* logits = out + (size_t)t * V_;
    gemm(hbf, wout_bf, b_out, logits, B_, V_, H_, H_, H_, (long)(T_ - 1) * V_);
    k_logsoftmax<<<B_, 256, 0, stream>>>(logits, (long)(T_ - 1) * V_);
  }

  (void)in_sizes; (void)n_in; (void)out_size; (void)ws_size;
}
